// DFD_10849087390476
// MI455X (gfx1250) — matrix operand order repaired and checked by reading, not compiled, not run on hardware
//
#include <hip/hip_runtime.h>

// ---- CDNA5 / gfx1250 WMMA types -------------------------------------------
typedef _Float16 v8h  __attribute__((ext_vector_type(8)));
typedef _Float16 v16h __attribute__((ext_vector_type(16)));
typedef float    v8f  __attribute__((ext_vector_type(8)));

#define HW       65536   // 256*256
#define NBATCH   4
#define NCH_OUT  279
#define HSTRIDE  72      // 64 channels + 8 pad: 16B alignment, conflict-free banks
#define WAVES    8

struct HeadW { const float *Win, *bin, *Wh, *bh, *Wout, *bout; };
struct KArgs { const float* x; float* out; HeadW h[5]; };

__device__ __forceinline__ v8f wmma_f16(v16h a, v16h b, v8f c) {
  // D = A(16x32 f16) * B(32x16 f16) + C(16x16 f32)
  return __builtin_amdgcn_wmma_f32_16x16x32_f16(false, a, false, b, (short)0, c, false, false);
}

__device__ __forceinline__ float relu1(float x) {
  // v_med3_f32(x, 0, +inf) == max(x, 0) in a single VALU op
  return __builtin_amdgcn_fmed3f(x, 0.0f, __builtin_inff());
}

// B fragment (32x16, f16): lane holds pixel N = lane&15; elems 0-7 -> K = kbase + (lane>>4)*8 + e,
// elems 8-15 -> K = kbase + 16 + (lane>>4)*8 + (e-8).
__device__ __forceinline__ v16h load_bfrag(const _Float16* hb, int lane, int kbase) {
  const int pix = lane & 15;
  const int c0  = kbase + ((lane >> 4) << 3);
  const _Float16* p = hb + pix * HSTRIDE + c0;
  v8h lo = *(const v8h*)(p);
  v8h hi = *(const v8h*)(p + 16);
  return __builtin_shufflevector(lo, hi, 0,1,2,3,4,5,6,7,8,9,10,11,12,13,14,15);
}

// A fragment (16x32, f16) from padded row-major [O][HSTRIDE] weights:
// lane row M = mt*16 + (lane&15); elems 0-7 -> K = kbase + (lane>>4)*8 + e,
// elems 8-15 -> K = kbase + 16 + (lane>>4)*8 + (e-8). (ISA 16-bit A 16x32 layout)
__device__ __forceinline__ v16h load_afrag(const _Float16* w, int lane, int mt, int kbase) {
  const int r  = mt * 16 + (lane & 15);
  const int k0 = kbase + ((lane >> 4) << 3);
  const _Float16* p = w + r * HSTRIDE;
  v8h lo = *(const v8h*)(p + k0);
  v8h hi = *(const v8h*)(p + k0 + 16);
  return __builtin_shufflevector(lo, hi, 0,1,2,3,4,5,6,7,8,9,10,11,12,13,14,15);
}

// One 64->64 projection layer, in-place on the wave's LDS activation tile.
// All 4 M-tile WMMA chains are issued before any epilogue so the scheduler can
// hide the WMMA->VALU hazard window with the next tile's WMMAs.
// D-layout: VGPR v, lane L -> channel = mt*16 + v + 8*(L>>4), pixel = L&15.
template<bool SKIP>
__device__ __forceinline__ void proj64(const _Float16* W, const float* bias,
                                       _Float16* hb, int lane) {
  const int pix   = lane & 15;
  const int half8 = (lane >> 4) << 3;
  // Read the full 64-channel B operand before any in-place store.
  v16h b0 = load_bfrag(hb, lane, 0);
  v16h b1 = load_bfrag(hb, lane, 32);

  v8f acc[4];
#pragma unroll
  for (int mt = 0; mt < 4; ++mt) {
    const int chb = mt * 16 + half8;
#pragma unroll
    for (int v = 0; v < 8; ++v) acc[mt][v] = bias[chb + v];
  }
#pragma unroll
  for (int mt = 0; mt < 4; ++mt) {
    v16h a0 = load_afrag(W, lane, mt, 0);
    v16h a1 = load_afrag(W, lane, mt, 32);
    acc[mt] = wmma_f16(a0, b0, acc[mt]);
    acc[mt] = wmma_f16(a1, b1, acc[mt]);
  }
#pragma unroll
  for (int mt = 0; mt < 4; ++mt) {
    const int chb = mt * 16 + half8;
    _Float16* dst = hb + pix * HSTRIDE + chb;
    v8h ho;
    if (SKIP) ho = *(const v8h*)dst;          // residual (old h)
    v8h o;
#pragma unroll
    for (int v = 0; v < 8; ++v) o[v] = (_Float16)relu1(acc[mt][v]);
    if (SKIP) o = o + ho;                     // v_pk_add_f16 x4
    *(v8h*)dst = o;
  }
}

__global__ __launch_bounds__(256) void dfd_heads(KArgs args) {
  __shared__ __align__(16) _Float16 sWin [64  * HSTRIDE];
  __shared__ __align__(16) _Float16 sWh  [64  * HSTRIDE];
  __shared__ __align__(16) _Float16 sWout[128 * HSTRIDE];
  __shared__ __align__(16) _Float16 sH   [WAVES * 16 * HSTRIDE];
  __shared__ __align__(32) float sBin[64], sBh[128], sBout[128];

  const int tid   = threadIdx.x;
  const int lane  = tid & 31;
  const int wave  = tid >> 5;
  const int pix   = lane & 15;
  const int half8 = (lane >> 4) << 3;

  const int tile = blockIdx.x;              // 2048 blocks: 512 per batch image
  const int b    = tile >> 9;
  const int p0   = (tile & 511) * (WAVES * 16) + wave * 16;

  _Float16* hb = &sH[wave * 16 * HSTRIDE];

  const int odArr[5]  = {81, 2, 4, 64, 128};
  const int offArr[5] = { 0, 81, 83, 87, 151};

  for (int t = 0; t < 5; ++t) {
    const HeadW hw = args.h[t];
    const int   od = odArr[t];

    __syncthreads();  // previous head done with shared weights
    for (int i = tid; i < 64 * 64; i += 256)
      sWin[(i >> 6) * HSTRIDE + (i & 63)] = (_Float16)hw.Win[i];
    for (int i = tid; i < 64 * 64; i += 256)
      sWh[(i >> 6) * HSTRIDE + (i & 63)] = (_Float16)hw.Wh[i];            // hidden layer 0
    for (int i = tid; i < od * 64; i += 256)
      sWout[(i >> 6) * HSTRIDE + (i & 63)] = (_Float16)hw.Wout[i];
    if (tid < 64)  sBin[tid]  = hw.bin[tid];
    if (tid < 128) sBh[tid]   = hw.bh[tid];
    if (tid < od)  sBout[tid] = hw.bout[tid];
    __syncthreads();

    // Load x tile (f32 global, channel-major) -> f16 LDS [pix][ch]
    {
      const float* xb = args.x + (size_t)b * 64 * HW + p0;
#pragma unroll 8
      for (int i = 0; i < 32; ++i) {
        const int ch = i * 2 + (lane >> 4);
        hb[pix * HSTRIDE + ch] = (_Float16)xb[(size_t)ch * HW + pix];
      }
    }

    proj64<false>(sWin, sBin, hb, lane);       // in-layer: relu(Win x + bin)
    proj64<true >(sWh,  sBh,  hb, lane);       // hidden 0: h += relu(Wh0 h + bh0)

    __syncthreads();                           // all waves done reading Wh layer 0
    for (int i = tid; i < 64 * 64; i += 256)
      sWh[(i >> 6) * HSTRIDE + (i & 63)] = (_Float16)hw.Wh[64 * 64 + i];  // hidden layer 1
    __syncthreads();

    proj64<true >(sWh, sBh + 64, hb, lane);    // hidden 1

    // Out layer: Wout h + bout -> global. Tiles processed in pairs (two
    // independent accumulator chains) to hide WMMA hazards; channel-tail masked.
    {
      v16h b0 = load_bfrag(hb, lane, 0);
      v16h b1 = load_bfrag(hb, lane, 32);
      float* ob = args.out + (size_t)b * NCH_OUT * HW + (size_t)offArr[t] * HW + p0;
      const int ntiles = (od + 15) >> 4;
      for (int mt = 0; mt < ntiles; mt += 2) {
        const bool has2 = (mt + 1) < ntiles;   // uniform across the wave
        const int  mtB  = has2 ? mt + 1 : mt;

        v16h a00 = load_afrag(sWout, lane, mt,  0);
        v16h a01 = load_afrag(sWout, lane, mt,  32);
        v16h a10 = load_afrag(sWout, lane, mtB, 0);
        v16h a11 = load_afrag(sWout, lane, mtB, 32);

        const int chbA = mt  * 16 + half8;
        const int chbB = mtB * 16 + half8;
        v8f accA, accB;
#pragma unroll
        for (int v = 0; v < 8; ++v) {
          const int cA = chbA + v, cB = chbB + v;
          accA[v] = (cA < od) ? sBout[cA] : 0.0f;
          accB[v] = (cB < od) ? sBout[cB] : 0.0f;
        }
        accA = wmma_f16(a00, b0, accA);
        accB = wmma_f16(a10, b0, accB);
        accA = wmma_f16(a01, b1, accA);
        accB = wmma_f16(a11, b1, accB);

#pragma unroll
        for (int v = 0; v < 8; ++v) {
          const int cA = chbA + v;
          if (cA < od) ob[(size_t)cA * HW + pix] = accA[v];
        }
        if (has2) {
#pragma unroll
          for (int v = 0; v < 8; ++v) {
            const int cB = chbB + v;
            if (cB < od) ob[(size_t)cB * HW + pix] = accB[v];
          }
        }
      }
    }
  }
}

extern "C" void kernel_launch(void* const* d_in, const int* in_sizes, int n_in,
                              void* d_out, int out_size, void* d_ws, size_t ws_size,
                              hipStream_t stream) {
  (void)in_sizes; (void)n_in; (void)d_ws; (void)ws_size; (void)out_size;
  KArgs a;
  a.x   = (const float*)d_in[0];
  a.out = (float*)d_out;
  for (int t = 0; t < 5; ++t) {
    a.h[t].Win  = (const float*)d_in[1 + 6 * t + 0];
    a.h[t].bin  = (const float*)d_in[1 + 6 * t + 1];
    a.h[t].Wh   = (const float*)d_in[1 + 6 * t + 2];
    a.h[t].bh   = (const float*)d_in[1 + 6 * t + 3];
    a.h[t].Wout = (const float*)d_in[1 + 6 * t + 4];
    a.h[t].bout = (const float*)d_in[1 + 6 * t + 5];
  }
  const int nblocks = NBATCH * HW / (WAVES * 16);  // 2048
  dfd_heads<<<nblocks, 256, 0, stream>>>(a);
}
